// Attention_50732153701013
// MI455X (gfx1250) — matrix operand order repaired and checked by reading, not compiled, not run on hardware
//
#include <hip/hip_runtime.h>

// ---------------------------------------------------------------------------
// Cross-attention (bf16 WMMA path) for MI455X / gfx1250, wave32.
//   B=2, N=2048, DIM=1024, HEADS=16, DH=64, SCALE=1/8
// Data movement: GLOBAL_LOAD_ASYNC_TO_LDS_B128 (ASYNCcnt) wherever the copy
// is a straight 16B-aligned memcpy; VGPR path only where we must convert
// (f32->bf16) or transpose.
// Workspace layout (bf16 elements), total ~75.5 MB:
//   [0)              Wqkv_bf   1024*3072
//   [3145728)        Wout_bf   1024*1024
//   [4194304)        qkv       2(src)*3(q/k/v)*2(b)*16(h)*2048(n)*64(d)
//   [29360128)       attn_out  2(src)*2(b)*2048(n)*1024
// ---------------------------------------------------------------------------

typedef __bf16 bf16_t;
typedef __attribute__((ext_vector_type(16))) __bf16 v16bf;
typedef __attribute__((ext_vector_type(8)))  __bf16 v8bf;
typedef __attribute__((ext_vector_type(8)))  float  v8f;

#define HEADS 16
#define DH 64
#define NSEQ 2048
#define DIM 1024
#define QKV_COLS 3072
#define SCALE 0.125f

// ---- WMMA wrapper: D = A(16x32 bf16) * B(32x16 bf16) + C(16x16 f32) -------
__device__ __forceinline__ v8f wmma_bf16(v16bf a, v16bf b, v8f c) {
  return __builtin_amdgcn_wmma_f32_16x16x32_bf16(
      /*neg_a=*/false, a, /*neg_b=*/false, b,
      /*c_mod=*/(short)0, c, /*reuse_a=*/false, /*reuse_b=*/false);
}

__device__ __forceinline__ v16bf cat8(v8bf lo, v8bf hi) {
  return __builtin_shufflevector(lo, hi, 0, 1, 2, 3, 4, 5, 6, 7,
                                 8, 9, 10, 11, 12, 13, 14, 15);
}

// ---- async global -> LDS, 16 bytes per lane (GLOBAL_LOAD_ASYNC_TO_LDS_B128)
// GV mode: 64-bit global address in a VGPR pair; LDS byte address in a VGPR
// (low 32 bits of the generic pointer to a __shared__ object are the
// wave-relative LDS byte address per the flat-aperture mapping).
__device__ __forceinline__ void async_ld16(void* lds, const void* g) {
  unsigned l = (unsigned)(unsigned long long)(uintptr_t)lds;
  unsigned long long a = (unsigned long long)(uintptr_t)g;
  asm volatile("global_load_async_to_lds_b128 %0, %1, off"
               :: "v"(l), "v"(a) : "memory");
}
__device__ __forceinline__ void wait_async0() {
  asm volatile("s_wait_asynccnt 0x0" ::: "memory");
}

// A fragment (16x32, M x K), ISA 7.12.2 16-bit A layout.
// p points at (row0, col0) of a row-major LDS tile with given pitch (bf16).
// lane: m = lane&15 ; element e -> k = (e/8)*16 + (lane>>4)*8 + e%8
__device__ __forceinline__ v16bf frag_a(const bf16_t* p, int pitch) {
  const int lane = threadIdx.x & 31;
  const bf16_t* r = p + (lane & 15) * pitch + ((lane >> 4) * 8);
  v8bf lo = *(const v8bf*)r;         // k = half*8 + 0..7
  v8bf hi = *(const v8bf*)(r + 16);  // k = 16 + half*8 + 0..7
  return cat8(lo, hi);
}

// B fragment (32x16, K x N) read from an LDS tile stored [n][k] (N-major).
// lane: n = lane&15 ; element e -> k = (e/8)*16 + (lane>>4)*8 + e%8
__device__ __forceinline__ v16bf frag_b(const bf16_t* p, int pitch) {
  const int lane = threadIdx.x & 31;
  const bf16_t* r = p + (lane & 15) * pitch + ((lane >> 4) * 8);
  v8bf lo = *(const v8bf*)r;
  v8bf hi = *(const v8bf*)(r + 16);
  return cat8(lo, hi);
}

// ---------------------------------------------------------------------------
// Kernel 0: f32 -> bf16 conversion (weights)
// ---------------------------------------------------------------------------
__global__ __launch_bounds__(256) void cvt_kernel(const float* __restrict__ src,
                                                  bf16_t* __restrict__ dst, int n) {
  int i = blockIdx.x * 256 + threadIdx.x;
  if (i < n) dst[i] = (bf16_t)src[i];
}

// ---------------------------------------------------------------------------
// Kernel 1: QKV GEMM.  [4096,1024](f32, per src) x [1024,3072](bf16)
// 128x128 tile / block (256 thr = 8 waves, each wave 2x4 16x16 tiles).
// A stage must convert f32->bf16, B stage must transpose -> VGPR path here.
// Epilogue scatters into qkv[src][which][b][h][n][d] (bf16).
// ---------------------------------------------------------------------------
__global__ __launch_bounds__(256) void qkv_gemm_kernel(
    const float* __restrict__ x0, const float* __restrict__ x1,
    const bf16_t* __restrict__ Wb, bf16_t* __restrict__ qkv_out) {
  __shared__ alignas(16) bf16_t As[128][48];  // [m][k] row-major
  __shared__ alignas(16) bf16_t Bs[128][48];  // [n][k] (transposed on store)

  const int src = blockIdx.z;
  const float* __restrict__ A = src ? x1 : x0;
  const int mBase = blockIdx.y * 128;  // 0..4095
  const int nBase = blockIdx.x * 128;  // 0..3071
  const int t = threadIdx.x;
  const int lane = t & 31, w = t >> 5;
  const int wr = w & 3, wc = w >> 2;

  v8f acc[2][4];
  for (int i = 0; i < 2; ++i)
    for (int j = 0; j < 4; ++j) acc[i][j] = (v8f){};

  for (int kt = 0; kt < DIM; kt += 32) {
    __syncthreads();
    {  // stage A: 128x32 f32 -> bf16; thread: row = t>>1, 16 cols
      const int r = t >> 1, c0 = (t & 1) * 16;
      const float* g = A + (size_t)(mBase + r) * DIM + kt + c0;
      for (int i = 0; i < 16; i += 4) {
        float4 f = *(const float4*)(g + i);
        As[r][c0 + i + 0] = (bf16_t)f.x;
        As[r][c0 + i + 1] = (bf16_t)f.y;
        As[r][c0 + i + 2] = (bf16_t)f.z;
        As[r][c0 + i + 3] = (bf16_t)f.w;
      }
    }
    {  // stage B transposed: 32(k) x 128(n); thread: k = t>>3, 16 n's
      const int k = t >> 3, n0 = (t & 7) * 16;
      const bf16_t* g = Wb + (size_t)(kt + k) * QKV_COLS + nBase + n0;
      v8bf u0 = *(const v8bf*)g;
      v8bf u1 = *(const v8bf*)(g + 8);
      for (int i = 0; i < 8; ++i) Bs[n0 + i][k] = u0[i];
      for (int i = 0; i < 8; ++i) Bs[n0 + 8 + i][k] = u1[i];
    }
    __syncthreads();

    v16bf af[2], bfr[4];
    for (int i = 0; i < 2; ++i) af[i] = frag_a(&As[(wr * 2 + i) * 16][0], 48);
    for (int j = 0; j < 4; ++j) bfr[j] = frag_b(&Bs[(wc * 4 + j) * 16][0], 48);
    for (int i = 0; i < 2; ++i)
      for (int j = 0; j < 4; ++j) acc[i][j] = wmma_bf16(af[i], bfr[j], acc[i][j]);
  }

  // epilogue: scatter into qkv[src][which][b][h][n][d]
  const int nl = lane & 15, half = lane >> 4;
  for (int i = 0; i < 2; ++i)
    for (int j = 0; j < 4; ++j)
      for (int v = 0; v < 8; ++v) {
        int m = mBase + (wr * 2 + i) * 16 + half * 8 + v;  // row in [0,4096)
        int c = nBase + (wc * 4 + j) * 16 + nl;            // col in [0,3072)
        int which = c >> 10, rem = c & 1023;
        int h = rem >> 6, d = rem & 63;
        int b = m >> 11, nrow = m & 2047;
        size_t idx =
            ((((size_t)(src * 3 + which) * 2 + b) * HEADS + h) * NSEQ + nrow) * DH + d;
        qkv_out[idx] = (bf16_t)acc[i][j][v];
      }
}

// ---------------------------------------------------------------------------
// Kernel 2: flash-style cross attention.
// Block = (q-tile of 128, b*h, src).  8 waves, wave w owns 16 query rows.
// Q and K tiles staged with async-to-LDS; V needs a transpose (VGPR path).
// Streams 64-key chunks: S = Q K'^T (WMMA) -> online softmax (f32) ->
// P through LDS -> O += P V' (WMMA).
// ---------------------------------------------------------------------------
__global__ __launch_bounds__(256) void attn_kernel(
    const bf16_t* __restrict__ qkv, bf16_t* __restrict__ attn_out) {
  __shared__ alignas(16) bf16_t Qs[128][72];     // [q][d]
  __shared__ alignas(16) bf16_t Ks[64][72];      // [key][d]   == [n][k] for S
  __shared__ alignas(16) bf16_t Vt[64][72];      // [d][key]   == [n][k] for PV
  __shared__ alignas(16) bf16_t Ps[8][16][72];   // per-wave P [m][key]

  const int qt = blockIdx.x;                 // 0..15
  const int b = blockIdx.y >> 4, h = blockIdx.y & 15;
  const int src = blockIdx.z, osrc = 1 - src;
  const int t = threadIdx.x, lane = t & 31, w = t >> 5;
  const int nl = lane & 15, half = lane >> 4;

  const size_t headSz = (size_t)NSEQ * DH;
  const bf16_t* Qg = qkv + (((size_t)(src * 3 + 0) * 2 + b) * HEADS + h) * headSz;
  const bf16_t* Kg = qkv + (((size_t)(osrc * 3 + 1) * 2 + b) * HEADS + h) * headSz;
  const bf16_t* Vg = qkv + (((size_t)(osrc * 3 + 2) * 2 + b) * HEADS + h) * headSz;

  {  // stage Q tile (128 x 64), once, async global->LDS
    const int r = t >> 1, seg = (t & 1) * 32;
    const bf16_t* g = Qg + (size_t)(qt * 128 + r) * DH + seg;
    for (int i = 0; i < 32; i += 8)
      async_ld16(&Qs[r][seg + i], g + i);
  }
  wait_async0();
  __syncthreads();

  v16bf qa[2];
  qa[0] = frag_a(&Qs[w * 16][0], 72);
  qa[1] = frag_a(&Qs[w * 16][32], 72);

  v8f o[4];
  for (int nt = 0; nt < 4; ++nt) o[nt] = (v8f){};
  float rowmax[8], rowsum[8];
  for (int v = 0; v < 8; ++v) { rowmax[v] = -1e30f; rowsum[v] = 0.f; }

  for (int kc = 0; kc < NSEQ; kc += 64) {
    __syncthreads();  // all waves done with previous K/V chunk
    {  // stage K chunk (64 x 64) row-major: async global->LDS
      const int r = t >> 2, seg = (t & 3) * 16;
      const bf16_t* g = Kg + (size_t)(kc + r) * DH + seg;
      async_ld16(&Ks[r][seg], g);
      async_ld16(&Ks[r][seg + 8], g + 8);
    }
    {  // stage V chunk transposed -> Vt[d][key]  (VGPR path: transpose)
      const int r = t >> 2, seg = (t & 3) * 16;
      const bf16_t* g = Vg + (size_t)(kc + r) * DH + seg;
      v8bf u0 = *(const v8bf*)g;
      v8bf u1 = *(const v8bf*)(g + 8);
      for (int i = 0; i < 8; ++i) Vt[seg + i][r] = u0[i];
      for (int i = 0; i < 8; ++i) Vt[seg + 8 + i][r] = u1[i];
    }
    wait_async0();
    __syncthreads();

    // ---- S = Q * K^T (16 x 64 per wave) ----
    v8f s[4];
    for (int st = 0; st < 4; ++st) {
      s[st] = (v8f){};
      s[st] = wmma_bf16(qa[0], frag_b(&Ks[st * 16][0], 72), s[st]);
      s[st] = wmma_bf16(qa[1], frag_b(&Ks[st * 16][32], 72), s[st]);
    }

    // ---- online softmax over this 64-key chunk ----
    for (int st = 0; st < 4; ++st)
      for (int v = 0; v < 8; ++v) s[st][v] *= SCALE;

    float cm[8];
    for (int v = 0; v < 8; ++v) {
      float m = -1e30f;
      for (int st = 0; st < 4; ++st) m = fmaxf(m, s[st][v]);
      for (int msk = 1; msk < 16; msk <<= 1) m = fmaxf(m, __shfl_xor(m, msk, 16));
      cm[v] = m;
    }
    float corr[8], rs[8];
    for (int v = 0; v < 8; ++v) {
      float nm = fmaxf(rowmax[v], cm[v]);
      corr[v] = __expf(rowmax[v] - nm);
      rowmax[v] = nm;
      rs[v] = 0.f;
    }
    for (int st = 0; st < 4; ++st)
      for (int v = 0; v < 8; ++v) {
        float e = __expf(s[st][v] - rowmax[v]);
        s[st][v] = e;
        rs[v] += e;
      }
    for (int v = 0; v < 8; ++v) {
      float r = rs[v];
      for (int msk = 1; msk < 16; msk <<= 1) r += __shfl_xor(r, msk, 16);
      rowsum[v] = rowsum[v] * corr[v] + r;
    }
    for (int nt = 0; nt < 4; ++nt)
      for (int v = 0; v < 8; ++v) o[nt][v] *= corr[v];

    // ---- P -> per-wave LDS (lane-transpose), then O += P * V ----
    for (int st = 0; st < 4; ++st)
      for (int v = 0; v < 8; ++v)
        Ps[w][half * 8 + v][st * 16 + nl] = (bf16_t)s[st][v];
    asm volatile("s_wait_dscnt 0" ::: "memory");  // RAW on per-wave LDS region

    v16bf pa0 = frag_a(&Ps[w][0][0], 72);
    v16bf pa1 = frag_a(&Ps[w][0][32], 72);
    for (int nt = 0; nt < 4; ++nt) {
      o[nt] = wmma_bf16(pa0, frag_b(&Vt[nt * 16][0], 72), o[nt]);
      o[nt] = wmma_bf16(pa1, frag_b(&Vt[nt * 16][32], 72), o[nt]);
    }
  }

  // ---- normalize and store merged-heads output (bf16) ----
  float inv[8];
  for (int v = 0; v < 8; ++v) inv[v] = 1.f / rowsum[v];
  for (int nt = 0; nt < 4; ++nt)
    for (int v = 0; v < 8; ++v) {
      int m = qt * 128 + w * 16 + half * 8 + v;  // query row in [0,2048)
      int c = h * DH + nt * 16 + nl;             // merged col in [0,1024)
      size_t idx = (((size_t)src * 2 + b) * NSEQ + m) * DIM + c;
      attn_out[idx] = (bf16_t)(o[nt][v] * inv[v]);
    }
}

// ---------------------------------------------------------------------------
// Kernel 3: output projection. [8192,1024](bf16) x [1024,1024](bf16) + bias,
// f32 result straight into d_out (out then out1, matching row order).
// A tile staged async (straight bf16 copy); B tile transposed via VGPRs.
// ---------------------------------------------------------------------------
__global__ __launch_bounds__(256) void out_gemm_kernel(
    const bf16_t* __restrict__ Ab, const bf16_t* __restrict__ Wb,
    const float* __restrict__ bias, float* __restrict__ out) {
  __shared__ alignas(16) bf16_t As[128][48];
  __shared__ alignas(16) bf16_t Bs[128][48];

  const int mBase = blockIdx.y * 128;  // 0..8191
  const int nBase = blockIdx.x * 128;  // 0..1023
  const int t = threadIdx.x;
  const int lane = t & 31, w = t >> 5;
  const int wr = w & 3, wc = w >> 2;

  v8f acc[2][4];
  for (int i = 0; i < 2; ++i)
    for (int j = 0; j < 4; ++j) acc[i][j] = (v8f){};

  for (int kt = 0; kt < DIM; kt += 32) {
    __syncthreads();
    {  // stage A (bf16): async global->LDS, row = t>>1, 16 cols
      const int r = t >> 1, c0 = (t & 1) * 16;
      const bf16_t* g = Ab + (size_t)(mBase + r) * DIM + kt + c0;
      async_ld16(&As[r][c0], g);
      async_ld16(&As[r][c0 + 8], g + 8);
    }
    {  // stage B transposed (VGPR path)
      const int k = t >> 3, n0 = (t & 7) * 16;
      const bf16_t* g = Wb + (size_t)(kt + k) * DIM + nBase + n0;
      v8bf u0 = *(const v8bf*)g;
      v8bf u1 = *(const v8bf*)(g + 8);
      for (int i = 0; i < 8; ++i) Bs[n0 + i][k] = u0[i];
      for (int i = 0; i < 8; ++i) Bs[n0 + 8 + i][k] = u1[i];
    }
    wait_async0();
    __syncthreads();

    v16bf af[2], bfr[4];
    for (int i = 0; i < 2; ++i) af[i] = frag_a(&As[(wr * 2 + i) * 16][0], 48);
    for (int j = 0; j < 4; ++j) bfr[j] = frag_b(&Bs[(wc * 4 + j) * 16][0], 48);
    for (int i = 0; i < 2; ++i)
      for (int j = 0; j < 4; ++j) acc[i][j] = wmma_bf16(af[i], bfr[j], acc[i][j]);
  }

  const int nl = lane & 15, half = lane >> 4;
  for (int i = 0; i < 2; ++i)
    for (int j = 0; j < 4; ++j)
      for (int v = 0; v < 8; ++v) {
        int m = mBase + (wr * 2 + i) * 16 + half * 8 + v;
        int c = nBase + (wc * 4 + j) * 16 + nl;
        out[(size_t)m * DIM + c] = acc[i][j][v] + bias[c];
      }
}

// ---------------------------------------------------------------------------
extern "C" void kernel_launch(void* const* d_in, const int* in_sizes, int n_in,
                              void* d_out, int out_size, void* d_ws, size_t ws_size,
                              hipStream_t stream) {
  const float* x    = (const float*)d_in[0];  // (2,2048,1024)
  const float* x1   = (const float*)d_in[1];  // (2,2048,1024)
  const float* Wqkv = (const float*)d_in[2];  // (1024,3072)
  const float* Wout = (const float*)d_in[3];  // (1024,1024)
  const float* bout = (const float*)d_in[4];  // (1024,)

  bf16_t* ws      = (bf16_t*)d_ws;
  bf16_t* wqkv_b  = ws;                                   // 3,145,728 elems
  bf16_t* wout_b  = wqkv_b + (size_t)DIM * QKV_COLS;      // 1,048,576 elems
  bf16_t* qkvbuf  = wout_b + (size_t)DIM * DIM;           // 25,165,824 elems
  bf16_t* attnbuf = qkvbuf + (size_t)2 * 3 * 2 * HEADS * NSEQ * DH;  // 8,388,608

  (void)in_sizes; (void)n_in; (void)out_size; (void)ws_size;

  cvt_kernel<<<(DIM * QKV_COLS + 255) / 256, 256, 0, stream>>>(Wqkv, wqkv_b,
                                                               DIM * QKV_COLS);
  cvt_kernel<<<(DIM * DIM + 255) / 256, 256, 0, stream>>>(Wout, wout_b, DIM * DIM);

  // QKV GEMM: grid = (N tiles 3072/128, M tiles 4096/128, src)
  qkv_gemm_kernel<<<dim3(24, 32, 2), 256, 0, stream>>>(x, x1, wqkv_b, qkvbuf);

  // Attention: grid = (q tiles 2048/128, b*h, src)
  attn_kernel<<<dim3(16, 32, 2), 256, 0, stream>>>(qkvbuf, attnbuf);

  // Output projection: grid = (N tiles 1024/128, M tiles 8192/128)
  out_gemm_kernel<<<dim3(8, 64), 256, 0, stream>>>(attnbuf, wout_b, bout,
                                                   (float*)d_out);
}
